// Transformer_71485435674947
// MI455X (gfx1250) — matrix operand order repaired and checked by reading, not compiled, not run on hardware
//
#include <hip/hip_runtime.h>

// ---------------------------------------------------------------------------
// MI455X (gfx1250) transformer block.
// - bf16 WMMA (v_wmma_f32_16x16x32_bf16), fp32 accumulate
// - async global->LDS staging (global_load_async_to_lds_b128 + s_wait_asynccnt)
// - fragment batching so WMMA chains run without per-op dscnt stalls
// - wave32, 256-thread blocks = 8 waves
// ---------------------------------------------------------------------------

typedef __attribute__((ext_vector_type(16))) __bf16 v16bf;
typedef __attribute__((ext_vector_type(8)))  float  v8f;

struct alignas(16) U128 { unsigned int w[4]; };

constexpr int D_MODEL = 512;
constexpr int D_FF    = 2048;
constexpr int SEQ     = 2048;
constexpr int BATCH   = 8;
constexpr int NTOK    = BATCH * SEQ;           // 16384 token rows
constexpr float INV_SQRT_D = 0.04419417382415922f;  // 1/sqrt(512)

#define DEV static __device__ __forceinline__

DEV v8f wmma_bf16(v16bf a, v16bf b, v8f c) {
  return __builtin_amdgcn_wmma_f32_16x16x32_bf16(
      false, a, false, b, (short)0, c, false, false);
}

DEV v8f zero8() {
  v8f z;
#pragma unroll
  for (int i = 0; i < 8; ++i) z[i] = 0.f;
  return z;
}

// Wave-relative LDS byte offset of a __shared__ object (generic pointers to
// LDS carry the offset in the low 32 bits).
DEV unsigned lds_off(const void* p) { return (unsigned)(unsigned long long)p; }

// Async DMA: 16 bytes global -> LDS, tracked by ASYNCcnt.
DEV void async_copy16(unsigned lds_byte, const void* g) {
  asm volatile("global_load_async_to_lds_b128 %0, %1, off"
               :: "v"(lds_byte), "v"((unsigned long long)g)
               : "memory");
}
DEV void wait_async0() { asm volatile("s_wait_asynccnt 0x0" ::: "memory"); }

// A-matrix fragment (16x32 bf16, row-major source, stride ld elements).
// Lanes 0-15: K = {k0..k0+7, k0+16..k0+23}; lanes 16-31: +8.
DEV v16bf load_fragA(const __bf16* p, int row, int ld, int k0) {
  const int lane = threadIdx.x & 31;
  const int kh = (lane & 16) ? 8 : 0;
  const __bf16* base = p + (size_t)row * ld + k0 + kh;
  v16bf r;
  ((U128*)&r)[0] = *(const U128*)(base);
  ((U128*)&r)[1] = *(const U128*)(base + 16);
  return r;
}

// B-matrix fragment (32x16 bf16). Source is B^T row-major (N x K, stride ld).
// Lanes 0-15: K = {k0..k0+7, k0+16..k0+23}; lanes 16-31: +8.
DEV v16bf load_fragB(const __bf16* p, int col, int ld, int k0) {
  const int lane = threadIdx.x & 31;
  const __bf16* base = p + (size_t)col * ld + k0 + ((lane & 16) ? 8 : 0);
  v16bf r;
  ((U128*)&r)[0] = *(const U128*)(base);
  ((U128*)&r)[1] = *(const U128*)(base + 16);
  return r;
}

// ---------------------------------------------------------------------------
// Weight convert+transpose: WT[n*K + k] = (bf16) W[k*N + n]
// ---------------------------------------------------------------------------
__global__ __launch_bounds__(256) void k_wT(const float* __restrict__ W,
                                            __bf16* __restrict__ WT,
                                            int K, int N) {
  size_t tid = (size_t)blockIdx.x * 256 + threadIdx.x;
  if (tid >= (size_t)K * N) return;
  int n = (int)(tid / K);
  int k = (int)(tid % K);
  WT[tid] = (__bf16)W[(size_t)k * N + n];
}

// ---------------------------------------------------------------------------
// Embedding + positional encoding -> bf16 activations
// ---------------------------------------------------------------------------
__global__ __launch_bounds__(256) void k_embed(const int* __restrict__ x,
                                               const float* __restrict__ emb,
                                               __bf16* __restrict__ hB) {
  size_t tid = (size_t)blockIdx.x * 256 + threadIdx.x;   // over NTOK*D
  int d = (int)(tid & (D_MODEL - 1));
  size_t tok = tid >> 9;
  int s = (int)(tok & (SEQ - 1));
  int id = x[tok];
  float v = emb[(size_t)id * D_MODEL + d];
  float angle = (float)s / powf(10000.0f, 2.0f * (float)d / (float)D_MODEL);
  v += (d & 1) ? cosf(angle) : sinf(angle);
  hB[tid] = (__bf16)v;
}

// ---------------------------------------------------------------------------
// bf16 WMMA GEMM: C[M,N] = A @ B + bias (optional ReLU).
// Grid (N/128, M/128), 8 waves; wave computes 16x128 (8 accumulators).
// B tile (128 cols x 32 k = 8 KB) double-buffered in LDS via async DMA.
// Per k-step: batch-load all 8 B fragments (one ds clause, one wait), then
// 8 back-to-back WMMAs. A operand register-prefetched one step ahead.
// ---------------------------------------------------------------------------
__global__ __launch_bounds__(256) void k_gemm(
    const __bf16* __restrict__ A, const __bf16* __restrict__ BT,
    const float* __restrict__ bias, int K, int N,
    float* __restrict__ Cf, __bf16* __restrict__ Cb,
    __bf16* __restrict__ CbT, int relu) {
  __shared__ __bf16 Bs[2][128 * 32];     // 2 x 8 KB stages

  const int w    = threadIdx.x >> 5;
  const int lane = threadIdx.x & 31;
  const int m0   = blockIdx.y * 128 + w * 16;
  const int n0   = blockIdx.x * 128;
  const int arow = m0 + (lane & 15);
  const int bcol = lane & 15;

  const unsigned bs_base = lds_off(&Bs[0][0]);

  // stage BT[n0..n0+127][k0..k0+31] into buffer `buf` (512 x 16B chunks)
  auto stage = [&](int k0, int buf) {
#pragma unroll
    for (int c = 0; c < 2; ++c) {
      int chunk = threadIdx.x + c * 256;        // 0..511
      int row = chunk >> 2;                     // 0..127  (n within tile)
      int sub = chunk & 3;                      // 4 x 8-element pieces
      async_copy16(bs_base + (unsigned)buf * 8192u + (unsigned)chunk * 16u,
                   BT + (size_t)(n0 + row) * K + k0 + sub * 8);
    }
  };

  v8f acc[8];
#pragma unroll
  for (int t = 0; t < 8; ++t) acc[t] = zero8();

  stage(0, 0);
  v16bf a_cur = load_fragA(A, arow, K, 0);
  wait_async0();
  __syncthreads();

  for (int k0 = 0; k0 < K; k0 += 32) {
    const int cur = (k0 >> 5) & 1;
    if (k0 + 32 < K) stage(k0 + 32, cur ^ 1);            // DMA next stage
    v16bf a_nxt = a_cur;
    if (k0 + 32 < K) a_nxt = load_fragA(A, arow, K, k0 + 32);

    const __bf16* bs = &Bs[cur][0];
    v16bf bfr[8];
#pragma unroll
    for (int t = 0; t < 8; ++t)
      bfr[t] = load_fragB(bs, t * 16 + bcol, 32, 0);     // ds clause
#pragma unroll
    for (int t = 0; t < 8; ++t)
      acc[t] = wmma_bf16(a_cur, bfr[t], acc[t]);         // back-to-back WMMA

    a_cur = a_nxt;
    wait_async0();
    __syncthreads();
  }

  const int nl = lane & 15;
  const int mo = (lane >> 4) << 3;
#pragma unroll
  for (int t = 0; t < 8; ++t) {
    int n = n0 + t * 16 + nl;
    float bn = bias[n];
#pragma unroll
    for (int r = 0; r < 8; ++r) {
      int m = m0 + mo + r;
      float v = acc[t][r] + bn;
      if (relu) v = fmaxf(v, 0.f);
      size_t idx = (size_t)m * N + n;
      if (Cf)  Cf[idx] = v;
      if (Cb)  Cb[idx] = (__bf16)v;
      if (CbT) {
        int bb = m / SEQ, sI = m % SEQ;
        CbT[((size_t)bb * N + n) * SEQ + sI] = (__bf16)v;
      }
    }
  }
}

// ---------------------------------------------------------------------------
// Flash-style attention. Workgroup = (batch, 16-query tile); 8 waves.
// Q tile (16x512) async-staged to LDS once; per 128-key block each wave does
// its 16-key score tile (WMMA), all 256 threads do the online-softmax stats,
// then each wave accumulates a 64-wide D slice of P@V (WMMA).
// ---------------------------------------------------------------------------
__global__ __launch_bounds__(256) void k_attn(const __bf16* __restrict__ qB,
                                              const __bf16* __restrict__ kB,
                                              const __bf16* __restrict__ vT,
                                              __bf16* __restrict__ ctxB) {
  const int b    = blockIdx.y;
  const int q0   = blockIdx.x * 16;
  const int w    = threadIdx.x >> 5;
  const int lane = threadIdx.x & 31;

  __shared__ __bf16 Ql[16 * D_MODEL];    // 16 KB
  __shared__ float  Sraw[16][128];       // 8 KB
  __shared__ __bf16 Pl[16][128];         // 4 KB
  __shared__ float  rowM[16], rowL[16], rowF[16];

  const __bf16* qb = qB + (size_t)b * SEQ * D_MODEL;
  const __bf16* kb = kB + (size_t)b * SEQ * D_MODEL;
  const __bf16* vt = vT + (size_t)b * D_MODEL * SEQ;

  // async-stage Q tile: 16 rows x 512 = 16 KB = 1024 x 16B chunks
  {
    unsigned qbase = lds_off(&Ql[0]);
#pragma unroll
    for (int c = 0; c < 4; ++c) {
      int chunk = threadIdx.x + c * 256;     // 0..1023
      int row = chunk >> 6;                  // 0..15
      int sub = chunk & 63;                  // 16B pieces within row
      async_copy16(qbase + (unsigned)chunk * 16u,
                   qb + (size_t)(q0 + row) * D_MODEL + sub * 8);
    }
  }
  if (threadIdx.x < 16) { rowM[threadIdx.x] = -3.0e38f; rowL[threadIdx.x] = 0.f; }
  wait_async0();
  __syncthreads();

  const int nl = lane & 15;
  const int mo = (lane >> 4) << 3;
  const int srow = threadIdx.x >> 4;       // softmax row (0..15)
  const int ssub = threadIdx.x & 15;       // 8 cols each

  v8f acc[4];
#pragma unroll
  for (int t = 0; t < 4; ++t) acc[t] = zero8();

  for (int kb0 = 0; kb0 < SEQ; kb0 += 128) {
    // --- scores: S(16x16) = Q @ K^T for keys [kb0+w*16, +16) ---
    const int krow = kb0 + w * 16 + (lane & 15);
    v8f s = zero8();
    v16bf kf = load_fragB(kb, krow, D_MODEL, 0);
    v16bf qf = load_fragA(Ql, lane & 15, D_MODEL, 0);
    for (int k0 = 0; k0 < D_MODEL; k0 += 32) {
      v16bf kf_n = kf, qf_n = qf;
      if (k0 + 32 < D_MODEL) {
        kf_n = load_fragB(kb, krow, D_MODEL, k0 + 32);
        qf_n = load_fragA(Ql, lane & 15, D_MODEL, k0 + 32);
      }
      s = wmma_bf16(qf, kf, s);
      kf = kf_n;
      qf = qf_n;
    }
#pragma unroll
    for (int r = 0; r < 8; ++r)
      Sraw[mo + r][w * 16 + nl] = s[r] * INV_SQRT_D;
    __syncthreads();

    // --- online softmax: 16 threads per row, shfl reductions ---
    {
      float oldM = rowM[srow], oldL = rowL[srow];
      float mx = -3.0e38f;
#pragma unroll
      for (int j = 0; j < 8; ++j) mx = fmaxf(mx, Sraw[srow][ssub * 8 + j]);
#pragma unroll
      for (int m = 1; m < 16; m <<= 1) mx = fmaxf(mx, __shfl_xor(mx, m, 32));
      float newM = fmaxf(oldM, mx);
      float sum = 0.f;
#pragma unroll
      for (int j = 0; j < 8; ++j) {
        float e = __expf(Sraw[srow][ssub * 8 + j] - newM);
        Pl[srow][ssub * 8 + j] = (__bf16)e;
        sum += e;
      }
#pragma unroll
      for (int m = 1; m < 16; m <<= 1) sum += __shfl_xor(sum, m, 32);
      if (ssub == 0) {
        float f = __expf(oldM - newM);
        rowM[srow] = newM;
        rowL[srow] = oldL * f + sum;
        rowF[srow] = f;
      }
    }
    __syncthreads();

    // --- rescale running ctx, then ctx += P(16x128) @ V(128 x 64-slice) ---
    float fr[8];
#pragma unroll
    for (int r = 0; r < 8; ++r) fr[r] = rowF[mo + r];
#pragma unroll
    for (int t = 0; t < 4; ++t)
#pragma unroll
      for (int r = 0; r < 8; ++r) acc[t][r] *= fr[r];

    v16bf pf[4];
#pragma unroll
    for (int kk4 = 0; kk4 < 4; ++kk4)
      pf[kk4] = load_fragA(&Pl[0][0], lane & 15, 128, kk4 * 32);  // ds clause

    v16bf vf = load_fragB(vt, w * 64 + (lane & 15), SEQ, kb0);
#pragma unroll
    for (int idx = 0; idx < 16; ++idx) {
      const int kk4 = idx >> 2;
      const int t   = idx & 3;
      v16bf vf_n = vf;
      if (idx + 1 < 16) {
        const int nk = ((idx + 1) >> 2) << 5;
        const int nt = (idx + 1) & 3;
        vf_n = load_fragB(vt, w * 64 + nt * 16 + (lane & 15), SEQ, kb0 + nk);
      }
      acc[t] = wmma_bf16(pf[kk4], vf, acc[t]);
      vf = vf_n;
    }
    __syncthreads();
  }

  // --- normalize and store ctx (bf16) ---
  float invL[8];
#pragma unroll
  for (int r = 0; r < 8; ++r) invL[r] = 1.0f / rowL[mo + r];
#pragma unroll
  for (int t = 0; t < 4; ++t) {
#pragma unroll
    for (int r = 0; r < 8; ++r) {
      float v = acc[t][r] * invL[r];
      int dcol = w * 64 + t * 16 + nl;
      ctxB[((size_t)b * SEQ + q0 + mo + r) * D_MODEL + dcol] = (__bf16)v;
    }
  }
}

// ---------------------------------------------------------------------------
// Host driver
// ---------------------------------------------------------------------------
extern "C" void kernel_launch(void* const* d_in, const int* in_sizes, int n_in,
                              void* d_out, int out_size, void* d_ws, size_t ws_size,
                              hipStream_t stream) {
  (void)in_sizes; (void)n_in; (void)out_size; (void)ws_size;

  const int*   x   = (const int*)  d_in[0];
  const float* emb = (const float*)d_in[1];
  const float* wq  = (const float*)d_in[2];
  const float* bq  = (const float*)d_in[3];
  const float* wk  = (const float*)d_in[4];
  const float* bk  = (const float*)d_in[5];
  const float* wv  = (const float*)d_in[6];
  const float* bv  = (const float*)d_in[7];
  const float* wo  = (const float*)d_in[8];
  const float* bo  = (const float*)d_in[9];
  const float* w1  = (const float*)d_in[10];
  const float* b1  = (const float*)d_in[11];
  const float* w2  = (const float*)d_in[12];
  const float* b2  = (const float*)d_in[13];
  float* out = (float*)d_out;

  size_t off = 0;
  auto alloc = [&](size_t bytes) -> void* {
    off = (off + 255) & ~(size_t)255;
    void* p = (char*)d_ws + off;
    off += bytes;
    return p;
  };
  __bf16* hB   = (__bf16*)alloc((size_t)NTOK * D_MODEL * 2);  // h, reused as h2
  __bf16* qB   = (__bf16*)alloc((size_t)NTOK * D_MODEL * 2);
  __bf16* kBuf = (__bf16*)alloc((size_t)NTOK * D_MODEL * 2);
  __bf16* vT   = (__bf16*)alloc((size_t)BATCH * D_MODEL * SEQ * 2);
  __bf16* ctxB = (__bf16*)alloc((size_t)NTOK * D_MODEL * 2);
  __bf16* ffB  = (__bf16*)alloc((size_t)NTOK * D_FF * 2);
  __bf16* wqT  = (__bf16*)alloc((size_t)D_MODEL * D_MODEL * 2);
  __bf16* wkT  = (__bf16*)alloc((size_t)D_MODEL * D_MODEL * 2);
  __bf16* wvT  = (__bf16*)alloc((size_t)D_MODEL * D_MODEL * 2);
  __bf16* woT  = (__bf16*)alloc((size_t)D_MODEL * D_MODEL * 2);
  __bf16* w1T  = (__bf16*)alloc((size_t)D_MODEL * D_FF * 2);
  __bf16* w2T  = (__bf16*)alloc((size_t)D_FF * D_MODEL * 2);

  {
    int n = D_MODEL * D_MODEL;
    int g = (n + 255) / 256;
    k_wT<<<g, 256, 0, stream>>>(wq, wqT, D_MODEL, D_MODEL);
    k_wT<<<g, 256, 0, stream>>>(wk, wkT, D_MODEL, D_MODEL);
    k_wT<<<g, 256, 0, stream>>>(wv, wvT, D_MODEL, D_MODEL);
    k_wT<<<g, 256, 0, stream>>>(wo, woT, D_MODEL, D_MODEL);
    int n1 = D_MODEL * D_FF;
    int g1 = (n1 + 255) / 256;
    k_wT<<<g1, 256, 0, stream>>>(w1, w1T, D_MODEL, D_FF);
    k_wT<<<g1, 256, 0, stream>>>(w2, w2T, D_FF, D_MODEL);
  }

  {
    size_t ne = (size_t)NTOK * D_MODEL;
    k_embed<<<(unsigned)(ne / 256), 256, 0, stream>>>(x, emb, hB);
  }

  dim3 blk(256);
  dim3 gD(D_MODEL / 128, NTOK / 128);    // (4, 128)
  k_gemm<<<gD, blk, 0, stream>>>(hB, wqT, bq, D_MODEL, D_MODEL, nullptr, qB,   nullptr, 0);
  k_gemm<<<gD, blk, 0, stream>>>(hB, wkT, bk, D_MODEL, D_MODEL, nullptr, kBuf, nullptr, 0);
  k_gemm<<<gD, blk, 0, stream>>>(hB, wvT, bv, D_MODEL, D_MODEL, nullptr, nullptr, vT,  0);

  dim3 ga(SEQ / 16, BATCH);              // (128, 8)
  k_attn<<<ga, blk, 0, stream>>>(qB, kBuf, vT, ctxB);

  k_gemm<<<gD, blk, 0, stream>>>(ctxB, woT, bo, D_MODEL, D_MODEL, nullptr, hB, nullptr, 0);

  dim3 gF1(D_FF / 128, NTOK / 128);      // (16, 128)
  k_gemm<<<gF1, blk, 0, stream>>>(hB, w1T, b1, D_MODEL, D_FF, nullptr, ffB, nullptr, 1);
  dim3 gF2(D_MODEL / 128, NTOK / 128);   // (4, 128)
  k_gemm<<<gF2, blk, 0, stream>>>(ffB, w2T, b2, D_FF, D_MODEL, out, nullptr, nullptr, 0);
}
